// Wavenet_47201690583580
// MI455X (gfx1250) — matrix operand order repaired and checked by reading, compiled, not run on hardware
//
#include <hip/hip_runtime.h>
#include <hip/hip_bf16.h>
#include <math.h>

typedef __attribute__((ext_vector_type(16))) _Float16 v16h;
typedef __attribute__((ext_vector_type(8)))  _Float16 v8h;
typedef __attribute__((ext_vector_type(4)))  _Float16 h4;
typedef __attribute__((ext_vector_type(8)))  float    v8f;
typedef __attribute__((ext_vector_type(4)))  float    v4f;
typedef __attribute__((ext_vector_type(4)))  unsigned int v4u;
typedef __attribute__((ext_vector_type(4)))  int v4i;
typedef __attribute__((ext_vector_type(8)))  int v8i_t;

#define RC    128     // residual channels (WD)
#define SC    512     // skip channels (SD)
#define NB    8       // batch
#define TLEN  8192
#define FOUT  6145    // T - (FIELD-1)
#define SKIP0 2047    // first original-time index contributing to skip
#define MT    32      // time-tile rows per workgroup
#define NLAY  20

#if __has_builtin(__builtin_amdgcn_tensor_load_to_lds) && __has_builtin(__builtin_amdgcn_s_wait_tensorcnt)
#define HAS_TDM 1
#else
#define HAS_TDM 0
#endif

// ---------------------------------------------------------------------------
// WMMA fragment helpers (wave32, v_wmma_f32_16x16x32_f16 layouts per CDNA5 ISA)
//   A (16x32 f16): a[idx] = A[M=lane&15][K = (idx&7) | ((lane>=16)<<3) | ((idx>=8)<<4)]
//   B (32x16 f16) staged TRANSPOSED [N][32]: same K mapping as A per lane.
//   C/D (16x16 f32): c[j] = C[M = j + 8*(lane>=16)][N = lane&15]
// ---------------------------------------------------------------------------
__device__ __forceinline__ v16h frag_pair(const _Float16* p0, const _Float16* p1) {
  v8h lo = *(const v8h*)p0;
  v8h hi = *(const v8h*)p1;
  return __builtin_shufflevector(lo, hi, 0,1,2,3,4,5,6,7,8,9,10,11,12,13,14,15);
}

__device__ __forceinline__ v8f wmma16(v16h a, v16h b, v8f c) {
  return __builtin_amdgcn_wmma_f32_16x16x32_f16(false, a, false, b, (short)0, c,
                                                false, false);
}

#if HAS_TDM
// ---------------------------------------------------------------------------
// Tensor Data Mover: 2-D f16 tile (tile0 contiguous elems per row, tile1 rows,
// row stride in elems) global -> LDS (contiguous).  D# per CDNA5 ISA.
// This toolchain's builtin takes 6 args (g0 v4u, g1 v8i, g2 v4i, g3 v4i,
// extra v8i, cpol imm).
// ---------------------------------------------------------------------------
__device__ __forceinline__ void tdm_load_2d(unsigned lds_byte_addr, const void* gaddr,
                                            unsigned tile0, unsigned tile1,
                                            unsigned stride0) {
  unsigned long long ga = (unsigned long long)(size_t)gaddr;
  v4u g0;
  g0.x = 1u;                                   // count=1 (valid), load, user mode
  g0.y = lds_byte_addr;                        // lds_addr [63:32]
  g0.z = (unsigned)(ga & 0xffffffffu);         // global_addr [95:64]
  g0.w = (unsigned)((ga >> 32) & 0x01ffffffu) | (2u << 30);  // addr hi | type=2
  v8i_t g1;
  g1[0] = (int)(1u << 16);                     // wg_mask=0, data_size=1 (2 bytes)
  g1[1] = (int)((tile0 & 0xffffu) << 16);      // tensor_dim0[15:0] (== tile0)
  g1[2] = (int)((tile0 >> 16) | ((tile1 & 0xffffu) << 16));  // td0 hi | td1 lo
  g1[3] = (int)((tile1 >> 16) | ((tile0 & 0xffffu) << 16));  // td1 hi | tile_dim0
  g1[4] = (int)(tile1 & 0xffffu);              // tile_dim1 (tile_dim2 = 0)
  g1[5] = (int)stride0;                        // tensor_dim0_stride[31:0]
  g1[6] = 0;
  g1[7] = 0;
  v4i z4 = {0, 0, 0, 0};
  v8i_t z8 = {0, 0, 0, 0, 0, 0, 0, 0};
  __builtin_amdgcn_tensor_load_to_lds(g0, g1, z4, z4, z8, 0);
}
#endif

// Stage a [N][32] f16 B-chunk (K-chunk kc) from pre-transposed WT[N][K] into BT.
__device__ __forceinline__ void stage_weights(_Float16* BT, const _Float16* WT,
                                              int kc, int N, int K, int tid) {
#if HAS_TDM
  if ((tid >> 5) == 0) {
    tdm_load_2d((unsigned)(size_t)(void*)BT, WT + kc * 32, 32u, (unsigned)N,
                (unsigned)K);
    __builtin_amdgcn_s_wait_tensorcnt(0);
  }
#else
  for (int i = tid; i < N * 32; i += 256)
    BT[i] = WT[(size_t)(i >> 5) * K + kc * 32 + (i & 31)];
#endif
}

// ---------------------------------------------------------------------------
__global__ __launch_bounds__(RC)
void embed_kernel(const int* __restrict__ y, const float* __restrict__ emb,
                  float* __restrict__ x0, _Float16* __restrict__ xh0) {
  int row = blockIdx.x;
  int c = threadIdx.x;
  float v = emb[(size_t)y[row] * RC + c];
  x0[(size_t)row * RC + c] = v;
  xh0[(size_t)row * RC + c] = (_Float16)v;
}

// d[l][n*K+k] = (f16) s[l][k*N+n]   (transpose + convert, per-layer in blockIdx.y)
__global__ __launch_bounds__(256)
void cvtT16(const float* __restrict__ s, _Float16* __restrict__ d, int K, int N) {
  size_t base = (size_t)blockIdx.y * K * N;
  int o = blockIdx.x * 256 + threadIdx.x;
  if (o < K * N) {
    int n = o / K, k = o - n * K;
    d[base + o] = (_Float16)s[base + (size_t)k * N + n];
  }
}

__global__ __launch_bounds__(256)
void zero_f32(float* __restrict__ p, size_t n) {
  size_t i = (size_t)blockIdx.x * 256 + threadIdx.x;
  if (i < n) p[i] = 0.f;
}

// ---------------------------------------------------------------------------
// Fused WaveNet layer. x: fp32 [B,T,128] (residual master) + f16 mirror xh.
// ---------------------------------------------------------------------------
__global__ __launch_bounds__(256)
void wavenet_layer(const float* __restrict__ xin, float* __restrict__ xout,
                   const _Float16* __restrict__ xh, _Float16* __restrict__ xhout,
                   float* __restrict__ skip,
                   const _Float16* __restrict__ WdT,   // [256][256] (n,k)
                   const _Float16* __restrict__ WrT,   // [128][128]
                   const _Float16* __restrict__ WsT,   // [512][128]
                   const float* __restrict__ bd, const float* __restrict__ br,
                   const float* __restrict__ bs, int dil, int tbeg)
{
  __shared__ __align__(128) char smem[65536];
  _Float16* BT    = (_Float16*)smem;               // [N][32] weight chunk (<=32KB)
  _Float16* Acv   = (_Float16*)(smem + 32768);     // [2 taps][32][128] f16 (16KB)
  float*    inact = (float*)(smem + 32768);        // [32][256] fp32 (32KB), after conv
  _Float16* Ag    = (_Float16*)(smem + 32768);     // [32][128] gated f16, after gate

  const int tid  = threadIdx.x;
  const int lane = tid & 31;
  const int wid  = tid >> 5;
  const int b    = blockIdx.y;
  const int t0   = tbeg + blockIdx.x * MT;
  const int mi   = wid & 1;
  const int row  = mi * 16 + (lane & 15);
  const int hi8  = (lane >> 4) * 8;
  const int k16  = (lane >> 4) * 16;
  const int ncol = lane & 15;
  const int jrow = 8 * (lane >> 4);

  // ---- A tile: tap0 = xh[t-d] rows, tap1 = xh[t] rows (TDM 2-D copies) ----
#if HAS_TDM
  if (wid == 0) {
    unsigned a_off = (unsigned)(size_t)(void*)Acv;
    tdm_load_2d(a_off, xh + ((size_t)b * TLEN + (t0 - dil)) * RC, RC, MT, RC);
    tdm_load_2d(a_off + MT * RC * 2, xh + ((size_t)b * TLEN + t0) * RC, RC, MT, RC);
    __builtin_amdgcn_s_wait_tensorcnt(0);
  }
#else
  for (int i = tid; i < 2 * MT * RC; i += 256) {
    int tap = i >> 12, m = (i >> 7) & (MT - 1), c = i & (RC - 1);
    int t = t0 + m - (tap ? 0 : dil);
    _Float16 v = (_Float16)0.f;
    if (t0 + m < TLEN) v = xh[((size_t)b * TLEN + t) * RC + c];
    Acv[i] = v;
  }
#endif
  __syncthreads();

  // ---- GEMM1: in_act[32x256] = A[32x256] x Wd[256x256] ----
  v8f accC[4] = {};
  for (int kc = 0; kc < 8; ++kc) {
    __syncthreads();
    stage_weights(BT, WdT, kc, 256, 256, tid);
    __syncthreads();
    const _Float16* ap = Acv + (kc >> 2) * (MT * RC) + row * RC + (kc & 3) * 32 + hi8;
    v16h a = frag_pair(ap, ap + 16);
    for (int nt = 0; nt < 4; ++nt) {
      int n = (wid >> 1) * 64 + nt * 16 + ncol;
      const _Float16* bp = &BT[n * 32 + hi8];
      accC[nt] = wmma16(a, frag_pair(bp, bp + 16), accC[nt]);
    }
  }
  __syncthreads();                       // all Acv reads done before overwrite

  // ---- spill in_act to LDS (fp32) ----
  for (int nt = 0; nt < 4; ++nt) {
    int n = (wid >> 1) * 64 + nt * 16 + ncol;
    for (int j = 0; j < 8; ++j)
      inact[(mi * 16 + jrow + j) * 256 + n] = accC[nt][j];
  }
  __syncthreads();

  // ---- gate: tanh(.. + bd[:128]) * sigmoid(.. + bd[128:]) -> f16 Ag ----
  float g[16];
  for (int p = 0; p < 16; ++p) {
    int i = p * 256 + tid;               // enumerates (m, o), o fastest
    int m = i >> 7, o = i & (RC - 1);
    float ta = inact[m * 256 + o] + bd[o];
    float sg = inact[m * 256 + RC + o] + bd[RC + o];
    g[p] = tanhf(ta) * (1.f / (1.f + expf(-sg)));
  }
  __syncthreads();
  for (int p = 0; p < 16; ++p)
    Ag[p * 256 + tid] = (_Float16)g[p];  // Ag[m*128+o] == linear index
  __syncthreads();

  // ---- GEMM2: res[32x128] = Ag x Wr; + br + fp32 residual; refresh f16 mirror --
  v8f accR[2] = {};
  for (int kc = 0; kc < 4; ++kc) {
    __syncthreads();
    stage_weights(BT, WrT, kc, 128, 128, tid);
    __syncthreads();
    const _Float16* ap = &Ag[row * RC + kc * 32 + hi8];
    v16h a = frag_pair(ap, ap + 16);
    for (int nt = 0; nt < 2; ++nt) {
      int n = (wid >> 1) * 32 + nt * 16 + ncol;
      const _Float16* bp = &BT[n * 32 + hi8];
      accR[nt] = wmma16(a, frag_pair(bp, bp + 16), accR[nt]);
    }
  }
  for (int nt = 0; nt < 2; ++nt) {
    int n = (wid >> 1) * 32 + nt * 16 + ncol;
    for (int j = 0; j < 8; ++j) {
      int t = t0 + mi * 16 + jrow + j;
      if (t < TLEN) {
        size_t idx = ((size_t)b * TLEN + t) * RC + n;
        float v = accR[nt][j] + br[n] + xin[idx];   // x_new[t] = res + x_old[t]
        xout[idx] = v;
        xhout[idx] = (_Float16)v;
      }
    }
  }

  // ---- GEMM3: skip[32x512] += Ag x Ws + bs (rows with t >= 2047) ----
  if (t0 + MT > SKIP0) {                 // block-uniform
    v8f accS[8] = {};
    for (int kc = 0; kc < 4; ++kc) {
      __syncthreads();
      stage_weights(BT, WsT, kc, 512, 128, tid);
      __syncthreads();
      const _Float16* ap = &Ag[row * RC + kc * 32 + hi8];
      v16h a = frag_pair(ap, ap + 16);
      for (int nt = 0; nt < 8; ++nt) {
        int n = (wid >> 1) * 128 + nt * 16 + ncol;
        const _Float16* bp = &BT[n * 32 + hi8];
        accS[nt] = wmma16(a, frag_pair(bp, bp + 16), accS[nt]);
      }
    }
    for (int nt = 0; nt < 8; ++nt) {
      int n = (wid >> 1) * 128 + nt * 16 + ncol;
      for (int j = 0; j < 8; ++j) {
        int t = t0 + mi * 16 + jrow + j;
        if (t >= SKIP0 && t < TLEN) {
          size_t idx = ((size_t)b * FOUT + (t - SKIP0)) * SC + n;
          skip[idx] += accS[nt][j] + bs[n];
        }
      }
    }
  }
}

// ---------------------------------------------------------------------------
// post1: h1[r][512] = relu( relu(skip[r]) @ post1_w ) as f16, r = b*6145+k
// ---------------------------------------------------------------------------
__global__ __launch_bounds__(256)
void post1_kernel(const float* __restrict__ skip, const _Float16* __restrict__ WT,
                  _Float16* __restrict__ h1, int mtot)
{
  __shared__ __align__(128) char smem[65536];
  _Float16* BT = (_Float16*)smem;                 // [512][32] 32KB
  _Float16* Ah = (_Float16*)(smem + 32768);       // [32][512] 32KB

  const int tid  = threadIdx.x;
  const int lane = tid & 31;
  const int wid  = tid >> 5;
  const int mi   = wid & 1;
  const int row  = mi * 16 + (lane & 15);
  const int hi8  = (lane >> 4) * 8;
  const int k16  = (lane >> 4) * 16;
  const int ncol = lane & 15;
  const int jrow = 8 * (lane >> 4);
  const int r0   = blockIdx.x * MT;

  for (int i = tid; i < MT * SC / 4; i += 256) {   // float4 relu+convert
    int m = i >> 7, c4 = (i & 127) << 2;
    int r = r0 + m;
    v4f v = {};
    if (r < mtot) v = *(const v4f*)&skip[(size_t)r * SC + c4];
    v.x = v.x > 0.f ? v.x : 0.f;
    v.y = v.y > 0.f ? v.y : 0.f;
    v.z = v.z > 0.f ? v.z : 0.f;
    v.w = v.w > 0.f ? v.w : 0.f;
    *(h4*)&Ah[m * SC + c4] = __builtin_convertvector(v, h4);
  }
  __syncthreads();

  v8f acc[8] = {};
  for (int kc = 0; kc < 16; ++kc) {
    __syncthreads();
    stage_weights(BT, WT, kc, 512, 512, tid);
    __syncthreads();
    const _Float16* ap = &Ah[row * SC + kc * 32 + hi8];
    v16h a = frag_pair(ap, ap + 16);
    for (int nt = 0; nt < 8; ++nt) {
      int n = (wid >> 1) * 128 + nt * 16 + ncol;
      const _Float16* bp = &BT[n * 32 + hi8];
      acc[nt] = wmma16(a, frag_pair(bp, bp + 16), acc[nt]);
    }
  }
  for (int nt = 0; nt < 8; ++nt) {
    int n = (wid >> 1) * 128 + nt * 16 + ncol;
    for (int j = 0; j < 8; ++j) {
      int r = r0 + mi * 16 + jrow + j;
      if (r < mtot) {
        float v = acc[nt][j];
        h1[(size_t)r * SC + n] = (_Float16)(v > 0.f ? v : 0.f);
      }
    }
  }
}

// ---------------------------------------------------------------------------
// post2: out[b][o][k] = h1[r=b*6145+k] @ post2_w; LDS-transposed store
// ---------------------------------------------------------------------------
__global__ __launch_bounds__(256)
void post2_kernel(const _Float16* __restrict__ h1, const _Float16* __restrict__ WT,
                  float* __restrict__ out, int mtot)
{
  __shared__ __align__(128) char smem[49152];
  _Float16* BT = (_Float16*)smem;                 // [256][32] 16KB
  _Float16* Ah = (_Float16*)(smem + 16384);       // [32][512] 32KB
  float*    ob = (float*)(smem + 16384);          // [32][256] 32KB (reuses Ah)

  const int tid  = threadIdx.x;
  const int lane = tid & 31;
  const int wid  = tid >> 5;
  const int mi   = wid & 1;
  const int row  = mi * 16 + (lane & 15);
  const int hi8  = (lane >> 4) * 8;
  const int k16  = (lane >> 4) * 16;
  const int ncol = lane & 15;
  const int jrow = 8 * (lane >> 4);
  const int r0   = blockIdx.x * MT;

#if HAS_TDM
  if (wid == 0) {
    tdm_load_2d((unsigned)(size_t)(void*)Ah, h1 + (size_t)r0 * SC, SC, MT, SC);
    __builtin_amdgcn_s_wait_tensorcnt(0);
  }
#else
  for (int i = tid; i < MT * SC; i += 256) {
    int m = i >> 9, c = i & (SC - 1);
    int r = r0 + m;
    Ah[i] = (r < mtot) ? h1[(size_t)r * SC + c] : (_Float16)0.f;
  }
#endif
  __syncthreads();

  v8f acc[4] = {};
  for (int kc = 0; kc < 16; ++kc) {
    __syncthreads();
    stage_weights(BT, WT, kc, 256, 512, tid);
    __syncthreads();
    const _Float16* ap = &Ah[row * SC + kc * 32 + hi8];
    v16h a = frag_pair(ap, ap + 16);
    for (int nt = 0; nt < 4; ++nt) {
      int n = (wid >> 1) * 64 + nt * 16 + ncol;
      const _Float16* bp = &BT[n * 32 + hi8];
      acc[nt] = wmma16(a, frag_pair(bp, bp + 16), acc[nt]);
    }
  }
  __syncthreads();                        // Ah reads done; reuse as ob
  for (int nt = 0; nt < 4; ++nt) {
    int n = (wid >> 1) * 64 + nt * 16 + ncol;
    for (int j = 0; j < 8; ++j)
      ob[(mi * 16 + jrow + j) * 256 + n] = acc[nt][j];
  }
  __syncthreads();
  for (int i = tid; i < MT * 256; i += 256) {   // transpose store, contiguous in k
    int m = i & 31, o = i >> 5;
    int r = r0 + m;
    if (r < mtot) {
      int bb = r / FOUT;
      int k  = r - bb * FOUT;
      out[((size_t)bb * 256 + o) * FOUT + k] = ob[m * 256 + o];
    }
  }
}

// ---------------------------------------------------------------------------
extern "C" void kernel_launch(void* const* d_in, const int* in_sizes, int n_in,
                              void* d_out, int out_size, void* d_ws, size_t ws_size,
                              hipStream_t stream)
{
  (void)in_sizes; (void)n_in; (void)out_size; (void)ws_size;
  static const int DIL[NLAY] = {1,2,4,8,16,32,64,128,256,512,
                                1,2,4,8,16,32,64,128,256,512};
  const int*   y   = (const int*)d_in[0];
  const float* emb = (const float*)d_in[1];
  const float* Wd  = (const float*)d_in[2];
  const float* bd  = (const float*)d_in[3];
  const float* Wr  = (const float*)d_in[4];
  const float* br  = (const float*)d_in[5];
  const float* Ws  = (const float*)d_in[6];
  const float* bs  = (const float*)d_in[7];
  const float* p1  = (const float*)d_in[8];
  const float* p2  = (const float*)d_in[9];
  float* out = (float*)d_out;

  // Workspace: xA | xB | xhA | xhB | skip | transposed f16 weights. (~208 MB)
  // h1 (f16, 50MB) reuses the xA/xB region (67MB) after the layer loop.
  char* ws = (char*)d_ws;
  const size_t XB = (size_t)NB * TLEN * RC * sizeof(float);
  const size_t HB = (size_t)NB * TLEN * RC * sizeof(_Float16);
  const size_t SB = (size_t)NB * FOUT * SC * sizeof(float);
  float*    xA  = (float*)ws;
  float*    xB  = (float*)(ws + XB);
  _Float16* xhA = (_Float16*)(ws + 2 * XB);
  _Float16* xhB = (_Float16*)(ws + 2 * XB + HB);
  float*    skip = (float*)(ws + 2 * XB + 2 * HB);
  char* wb = ws + 2 * XB + 2 * HB + SB;
  _Float16* WdT = (_Float16*)wb;  wb += (size_t)NLAY * 256 * 256 * 2;
  _Float16* WrT = (_Float16*)wb;  wb += (size_t)NLAY * RC * RC * 2;
  _Float16* WsT = (_Float16*)wb;  wb += (size_t)NLAY * RC * SC * 2;
  _Float16* P1T = (_Float16*)wb;  wb += (size_t)SC * SC * 2;
  _Float16* P2T = (_Float16*)wb;
  _Float16* h1  = (_Float16*)ws;

  embed_kernel<<<NB * TLEN, RC, 0, stream>>>(y, emb, xA, xhA);

  // transpose+convert weights:  [K][N] -> [N][K] f16
  cvtT16<<<dim3(256, NLAY), 256, 0, stream>>>(Wd, WdT, 256, 256);  // K=2*RC, N=2*RC
  cvtT16<<<dim3(64,  NLAY), 256, 0, stream>>>(Wr, WrT, RC, RC);
  cvtT16<<<dim3(256, NLAY), 256, 0, stream>>>(Ws, WsT, RC, SC);
  cvtT16<<<dim3(1024, 1),   256, 0, stream>>>(p1, P1T, SC, SC);
  cvtT16<<<dim3(512, 1),    256, 0, stream>>>(p2, P2T, SC, 256);

  size_t sn = (size_t)NB * FOUT * SC;
  zero_f32<<<(int)((sn + 255) / 256), 256, 0, stream>>>(skip, sn);

  int start = 0;
  float* xin = xA;  float* xout = xB;
  _Float16* xh = xhA;  _Float16* xho = xhB;
  for (int i = 0; i < NLAY; ++i) {
    int d = DIL[i];
    int tbeg = start + d;
    int tiles = (TLEN - tbeg + MT - 1) / MT;
    wavenet_layer<<<dim3(tiles, NB), 256, 0, stream>>>(
        xin, xout, xh, xho, skip,
        WdT + (size_t)i * 256 * 256,
        WrT + (size_t)i * RC * RC,
        WsT + (size_t)i * RC * SC,
        bd + (size_t)i * 2 * RC, br + (size_t)i * RC, bs + (size_t)i * SC,
        d, tbeg);
    start += d;
    float* tf = xin; xin = xout; xout = tf;
    _Float16* th = xh; xh = xho; xho = th;
  }

  int mtot = NB * FOUT;                     // 49160 rows
  int gtiles = (mtot + MT - 1) / MT;        // 1537
  post1_kernel<<<gtiles, 256, 0, stream>>>(skip, P1T, h1, mtot);
  post2_kernel<<<gtiles, 256, 0, stream>>>(h1, P2T, out, mtot);
}
